// MultiHeadAttention_55387898249810
// MI455X (gfx1250) — matrix operand order repaired and checked by reading, not compiled, not run on hardware
//
#include <hip/hip_runtime.h>
#include <hip/hip_bf16.h>

typedef __attribute__((ext_vector_type(16))) __bf16 v16bf;
typedef __attribute__((ext_vector_type(8)))  __bf16 v8bf;
typedef __attribute__((ext_vector_type(8)))  float  v8f;

#define EMB 1024
#define NH 8
#define NKV 4
#define HD 128

static __device__ __forceinline__ v8f wmma_bf16(v16bf a, v16bf b, v8f c) {
  // 8 args: (neg_a, A, neg_b, B, c_mod, C, reuse_a, reuse_b)
  return __builtin_amdgcn_wmma_f32_16x16x32_bf16(false, a, false, b, (short)0, c, false, false);
}
static __device__ __forceinline__ v16bf cat16(v8bf lo, v8bf hi) {
  return __builtin_shufflevector(lo, hi, 0,1,2,3,4,5,6,7,8,9,10,11,12,13,14,15);
}

// ---- CDNA5 async global->LDS DMA (ASYNCcnt). INST_OFFSET applies to both
// ---- the LDS and the global address, so consecutive chunks share operands.
static __device__ __forceinline__ void async_ld32B(unsigned ldsaddr, const void* g) {
  asm volatile("global_load_async_to_lds_b128 %0, %1, off"
               :: "v"(ldsaddr), "v"(g) : "memory");
  asm volatile("global_load_async_to_lds_b128 %0, %1, off offset:16"
               :: "v"(ldsaddr), "v"(g) : "memory");
}
static __device__ __forceinline__ void async_ld64B(unsigned ldsaddr, const void* g) {
  asm volatile("global_load_async_to_lds_b128 %0, %1, off"
               :: "v"(ldsaddr), "v"(g) : "memory");
  asm volatile("global_load_async_to_lds_b128 %0, %1, off offset:16"
               :: "v"(ldsaddr), "v"(g) : "memory");
  asm volatile("global_load_async_to_lds_b128 %0, %1, off offset:32"
               :: "v"(ldsaddr), "v"(g) : "memory");
  asm volatile("global_load_async_to_lds_b128 %0, %1, off offset:48"
               :: "v"(ldsaddr), "v"(g) : "memory");
}
static __device__ __forceinline__ unsigned lds_addr(const void* p) {
  // flat shared address: low 32 bits are the LDS byte offset
  return (unsigned)(uintptr_t)p;
}

// ---------------- f32 -> bf16 conversion ----------------
__global__ __launch_bounds__(256) void f32_to_bf16_k(const float* __restrict__ in,
                                                     __bf16* __restrict__ out, int n) {
  int i = blockIdx.x * 256 + threadIdx.x;
  if (i < n) out[i] = (__bf16)in[i];
}

// ---------------- tiled bf16 WMMA GEMM: C[m,n] = sum_k A[m,k]*W[n,k] ----------------
// Double-buffered LDS, async-to-LDS staging, branch-free steady-state loop.
// mode 0: out bf16 [M,N]   mode 1: out bf16 transposed V -> [b][kv][d][T]   mode 2: out f32 [M,N]
__global__ __launch_bounds__(256) void gemm_bf16_wmma(
    const __bf16* __restrict__ A, const __bf16* __restrict__ W,
    void* __restrict__ out, int M, int N, int K, int mode, int Tdim)
{
  __shared__ __align__(16) __bf16 ldsA[2][128 * 40];
  __shared__ __align__(16) __bf16 ldsB[2][128 * 40];
  const int tid  = threadIdx.x;
  const int wave = tid >> 5, lane = tid & 31;
  const int nloc = lane & 15, kh = lane >> 4;
  const int m0 = blockIdx.y * 128, n0 = blockIdx.x * 128;

  const int row = tid >> 1, seg = tid & 1;
  const __bf16* gA = A + (size_t)(m0 + row) * K + seg * 16;
  const __bf16* gB = W + (size_t)(n0 + row) * K + seg * 16;
  unsigned aoff[2], boff[2];
  aoff[0] = lds_addr(&ldsA[0][row * 40 + seg * 16]);
  aoff[1] = lds_addr(&ldsA[1][row * 40 + seg * 16]);
  boff[0] = lds_addr(&ldsB[0][row * 40 + seg * 16]);
  boff[1] = lds_addr(&ldsB[1][row * 40 + seg * 16]);

  v8f acc[8] = {};
  const int nch = K >> 5;

  auto compute = [&](int cur) {
    int mrow = wave * 16 + nloc;
    v8bf alo = *(const v8bf*)&ldsA[cur][mrow * 40 + 8 * kh];
    v8bf ahi = *(const v8bf*)&ldsA[cur][mrow * 40 + 16 + 8 * kh];
    v16bf af = cat16(alo, ahi);
    #pragma unroll
    for (int j = 0; j < 8; ++j) {
      int nc = j * 16 + nloc;
      v8bf blo = *(const v8bf*)&ldsB[cur][nc * 40 + 8 * kh];
      v8bf bhi = *(const v8bf*)&ldsB[cur][nc * 40 + 16 + 8 * kh];
      acc[j] = wmma_bf16(af, cat16(blo, bhi), acc[j]);
    }
  };

  // prologue: stage chunk 0
  async_ld32B(aoff[0], gA);
  async_ld32B(boff[0], gB);

  // steady state: always stage next, keep only newest 4 loads in flight
  for (int c = 0; c < nch - 1; ++c) {
    const int cur = c & 1;
    async_ld32B(aoff[cur ^ 1], gA + (c + 1) * 32);
    async_ld32B(boff[cur ^ 1], gB + (c + 1) * 32);
    asm volatile("s_wait_asynccnt 0x4" ::: "memory");
    __syncthreads();
    compute(cur);
    __syncthreads();
  }
  // epilogue: last chunk
  asm volatile("s_wait_asynccnt 0x0" ::: "memory");
  __syncthreads();
  compute((nch - 1) & 1);

  #pragma unroll
  for (int j = 0; j < 8; ++j) {
    #pragma unroll
    for (int r = 0; r < 8; ++r) {
      int gm = m0 + wave * 16 + r + 8 * kh;
      int gn = n0 + j * 16 + nloc;
      float v = acc[j][r];
      if (mode == 0) {
        ((__bf16*)out)[(size_t)gm * N + gn] = (__bf16)v;
      } else if (mode == 1) {
        int b = gm / Tdim, t = gm - b * Tdim;
        int kvh = gn >> 7, d = gn & 127;
        ((__bf16*)out)[(((size_t)b * NKV + kvh) * HD + d) * Tdim + t] = (__bf16)v;
      } else {
        ((float*)out)[(size_t)gm * N + gn] = v;
      }
    }
  }
}

// ---------------- RoPE in place on bf16 [b,t,heads,128] ----------------
__global__ __launch_bounds__(256) void rope_bf16(__bf16* __restrict__ buf, int total,
                                                 int heads, int Tlen) {
  int gid = blockIdx.x * 256 + threadIdx.x;
  if (gid >= total) return;
  int i = gid & 63;
  int hd = gid >> 6;                 // (b*T + t)*heads + head
  int t = (hd / heads) % Tlen;
  size_t base = (size_t)hd * 128;
  float invf = __expf(-(float)i * (9.210340371976184f / 64.0f)); // 10000^(-i/64)
  float th = (float)t * invf;
  float s, c;
  __sincosf(th, &s, &c);
  float x1 = (float)buf[base + i];
  float x2 = (float)buf[base + 64 + i];
  buf[base + i]      = (__bf16)(x1 * c - x2 * s);
  buf[base + 64 + i] = (__bf16)(x1 * s + x2 * c);
}

// ---------------- flash attention: 4 waves x 16 q-rows, 32-wide s-chunks ----------------
// Double-buffered K/V tiles staged with async-to-LDS DMA.
__global__ __launch_bounds__(128) void attn_wmma(
    const __bf16* __restrict__ Qb,   // [b,t,h,128]  (== [b,t,1024])
    const __bf16* __restrict__ Kb,   // [b,s,kv,128] (== [b,s,512])
    const __bf16* __restrict__ Vt,   // [b,kv,128,T]
    __bf16* __restrict__ Ob,         // [b,t,h,128]
    int Tlen)
{
  __shared__ __align__(16) __bf16 ldsK[2][32 * 136];   // [s][d]
  __shared__ __align__(16) __bf16 ldsV[2][128 * 40];   // [d][s]
  __shared__ __align__(16) float  ldsS[4][16 * 36];    // per-wave S tile
  __shared__ __align__(16) __bf16 ldsP[4][16 * 40];    // per-wave P tile (bf16)
  __shared__ __align__(16) float  ldsRow[4][16];       // per-wave alpha / 1/l

  const int tid  = threadIdx.x;
  const int wave = tid >> 5, lane = tid & 31;
  const int nloc = lane & 15, kh = lane >> 4;
  const int b = blockIdx.z, h = blockIdx.y;
  const int kv = h >> 1;                 // GROUP = 2
  const int q0 = blockIdx.x * 64 + wave * 16;

  // preload Q A-fragments: 16 rows x 128 d
  v16bf qf[4];
  {
    const __bf16* qbase = Qb + ((size_t)(b * Tlen + q0 + nloc) * EMB + h * HD);
    #pragma unroll
    for (int dc = 0; dc < 4; ++dc) {
      v8bf lo = *(const v8bf*)(qbase + dc * 32 + 8 * kh);
      v8bf hi = *(const v8bf*)(qbase + dc * 32 + 16 + 8 * kh);
      qf[dc] = cat16(lo, hi);
    }
  }

  // per-thread tile staging addresses
  const int krow = tid >> 2, kd0 = (tid & 3) * 32;
  unsigned koff[2], voff[2];
  koff[0] = lds_addr(&ldsK[0][krow * 136 + kd0]);
  koff[1] = lds_addr(&ldsK[1][krow * 136 + kd0]);
  voff[0] = lds_addr(&ldsV[0][tid * 40]);
  voff[1] = lds_addr(&ldsV[1][tid * 40]);
  const __bf16* kbase = Kb + ((size_t)(b * Tlen + krow) * (NKV * HD) + kv * HD + kd0);
  const __bf16* vbase = Vt + (((size_t)b * NKV + kv) * HD + tid) * Tlen;

  v8f o[8] = {};
  float mrow = -1e30f, lrow = 0.f;

  auto stage = [&](int buf, int s0) {
    async_ld64B(koff[buf], kbase + (size_t)s0 * (NKV * HD));
    async_ld64B(voff[buf], vbase + s0);
  };

  auto body = [&](int cur, int s0) {
    // S = Q * K^T  (two 16x16 s-tiles, K-dim = 128 over 4 wmma each)
    v8f sacc[2] = {};
    #pragma unroll
    for (int dc = 0; dc < 4; ++dc) {
      #pragma unroll
      for (int st = 0; st < 2; ++st) {
        int sl = st * 16 + nloc;
        v8bf blo = *(const v8bf*)&ldsK[cur][sl * 136 + dc * 32 + 8 * kh];
        v8bf bhi = *(const v8bf*)&ldsK[cur][sl * 136 + dc * 32 + 16 + 8 * kh];
        sacc[st] = wmma_bf16(qf[dc], cat16(blo, bhi), sacc[st]);
      }
    }
    // scale + causal mask -> per-wave LDS S
    #pragma unroll
    for (int st = 0; st < 2; ++st) {
      #pragma unroll
      for (int r = 0; r < 8; ++r) {
        int mr = r + 8 * kh;
        float v = sacc[st][r] * 0.08838834764831845f; // 1/sqrt(128)
        if (s0 + st * 16 + nloc > q0 + mr) v = -1e30f;
        ldsS[wave][mr * 36 + st * 16 + nloc] = v;
      }
    }
    __syncthreads();

    // online softmax row pass (lanes 0..15 own one row each)
    if (lane < 16) {
      float rmax = -1e30f;
      #pragma unroll
      for (int j = 0; j < 32; ++j) rmax = fmaxf(rmax, ldsS[wave][lane * 36 + j]);
      float mnew = fmaxf(mrow, rmax);
      float al = __expf(mrow - mnew);
      float sum = 0.f;
      #pragma unroll
      for (int j = 0; j < 32; ++j) {
        float p = __expf(ldsS[wave][lane * 36 + j] - mnew);
        sum += p;
        ldsP[wave][lane * 40 + j] = (__bf16)p;
      }
      lrow = lrow * al + sum;
      mrow = mnew;
      ldsRow[wave][lane] = al;
    }
    __syncthreads();

    // rescale O, then O += P * V  (8 d-tiles, K-dim = 32)
    float alphas[8];
    #pragma unroll
    for (int r = 0; r < 8; ++r) alphas[r] = ldsRow[wave][r + 8 * kh];
    #pragma unroll
    for (int j = 0; j < 8; ++j)
      #pragma unroll
      for (int r = 0; r < 8; ++r) o[j][r] *= alphas[r];

    v8bf plo = *(const v8bf*)&ldsP[wave][nloc * 40 + 8 * kh];
    v8bf phi = *(const v8bf*)&ldsP[wave][nloc * 40 + 16 + 8 * kh];
    v16bf pf = cat16(plo, phi);
    #pragma unroll
    for (int dt = 0; dt < 8; ++dt) {
      int d = dt * 16 + nloc;
      v8bf vlo = *(const v8bf*)&ldsV[cur][d * 40 + 8 * kh];
      v8bf vhi = *(const v8bf*)&ldsV[cur][d * 40 + 16 + 8 * kh];
      o[dt] = wmma_bf16(pf, cat16(vlo, vhi), o[dt]);
    }
  };

  const int nchunks = blockIdx.x * 2 + 2;  // causal: cover s <= q0_block_end (>= 2)
  stage(0, 0);
  // steady state: stage next chunk (buffer c+1 == buffer c-1, freed by top barrier),
  // keep only the newest 8 per-thread async loads in flight.
  for (int c = 0; c < nchunks - 1; ++c) {
    const int cur = c & 1;
    __syncthreads();                 // prev iteration's consumers of buf cur^1 done
    stage(cur ^ 1, (c + 1) * 32);
    asm volatile("s_wait_asynccnt 0x8" ::: "memory");
    __syncthreads();                 // chunk c visible to all threads
    body(cur, c * 32);
  }
  // epilogue: last chunk
  __syncthreads();
  asm volatile("s_wait_asynccnt 0x0" ::: "memory");
  __syncthreads();
  body((nchunks - 1) & 1, (nchunks - 1) * 32);

  __syncthreads();
  if (lane < 16) ldsRow[wave][lane] = 1.f / lrow;
  __syncthreads();
  {
    float linv[8];
    #pragma unroll
    for (int r = 0; r < 8; ++r) linv[r] = ldsRow[wave][r + 8 * kh];
    #pragma unroll
    for (int dt = 0; dt < 8; ++dt) {
      #pragma unroll
      for (int r = 0; r < 8; ++r) {
        int t = q0 + r + 8 * kh;
        int d = dt * 16 + nloc;
        Ob[(size_t)(b * Tlen + t) * EMB + h * HD + d] = (__bf16)(o[dt][r] * linv[r]);
      }
    }
  }
}

// ---------------- host launch ----------------
extern "C" void kernel_launch(void* const* d_in, const int* in_sizes, int n_in,
                              void* d_out, int out_size, void* d_ws, size_t ws_size,
                              hipStream_t stream) {
  const float* x  = (const float*)d_in[0];
  const float* Wq = (const float*)d_in[1];
  const float* Wk = (const float*)d_in[2];
  const float* Wv = (const float*)d_in[3];
  const float* Wo = (const float*)d_in[4];

  const int Tlen = 2048;
  const int BT = in_sizes[0] / EMB;   // B*T = 8192
  const int B = BT / Tlen;

  char* ws = (char*)d_ws;
  __bf16* xbf = (__bf16*)ws; ws += (size_t)BT * EMB * 2;
  __bf16* wqb = (__bf16*)ws; ws += (size_t)EMB * EMB * 2;
  __bf16* wkb = (__bf16*)ws; ws += (size_t)(NKV * HD) * EMB * 2;
  __bf16* wvb = (__bf16*)ws; ws += (size_t)(NKV * HD) * EMB * 2;
  __bf16* wob = (__bf16*)ws; ws += (size_t)EMB * EMB * 2;
  __bf16* Qb  = (__bf16*)ws; ws += (size_t)BT * EMB * 2;
  __bf16* Kb  = (__bf16*)ws; ws += (size_t)BT * NKV * HD * 2;
  __bf16* Vt  = (__bf16*)ws; ws += (size_t)BT * NKV * HD * 2;
  __bf16* At  = (__bf16*)ws; ws += (size_t)BT * EMB * 2;

  f32_to_bf16_k<<<(BT * EMB + 255) / 256, 256, 0, stream>>>(x, xbf, BT * EMB);
  f32_to_bf16_k<<<(EMB * EMB + 255) / 256, 256, 0, stream>>>(Wq, wqb, EMB * EMB);
  f32_to_bf16_k<<<(NKV * HD * EMB + 255) / 256, 256, 0, stream>>>(Wk, wkb, NKV * HD * EMB);
  f32_to_bf16_k<<<(NKV * HD * EMB + 255) / 256, 256, 0, stream>>>(Wv, wvb, NKV * HD * EMB);
  f32_to_bf16_k<<<(EMB * EMB + 255) / 256, 256, 0, stream>>>(Wo, wob, EMB * EMB);

  dim3 blk(256);
  // Q = x @ Wq^T, K = x @ Wk^T, V^T built directly
  gemm_bf16_wmma<<<dim3(EMB / 128, BT / 128), blk, 0, stream>>>(xbf, wqb, Qb, BT, EMB, EMB, 0, Tlen);
  gemm_bf16_wmma<<<dim3((NKV * HD) / 128, BT / 128), blk, 0, stream>>>(xbf, wkb, Kb, BT, NKV * HD, EMB, 0, Tlen);
  gemm_bf16_wmma<<<dim3((NKV * HD) / 128, BT / 128), blk, 0, stream>>>(xbf, wvb, Vt, BT, NKV * HD, EMB, 1, Tlen);

  rope_bf16<<<(BT * NH * 64 + 255) / 256, 256, 0, stream>>>(Qb, BT * NH * 64, NH, Tlen);
  rope_bf16<<<(BT * NKV * 64 + 255) / 256, 256, 0, stream>>>(Kb, BT * NKV * 64, NKV, Tlen);

  attn_wmma<<<dim3(Tlen / 64, NH, B), dim3(128), 0, stream>>>(Qb, Kb, Vt, At, Tlen);

  // out = attn @ Wo^T (f32 output)
  gemm_bf16_wmma<<<dim3(EMB / 128, BT / 128), blk, 0, stream>>>(At, wob, d_out, BT, EMB, EMB, 2, Tlen);
}
